// LSTMLayer_24584392802363
// MI455X (gfx1250) — compile-verified, matrix operand order checked by reading, not run on hardware
//
#include <hip/hip_runtime.h>

// ---------------------------------------------------------------------------
// LSTM on gfx1250 (MI455X), bf16 WMMA path.
//   Stage 1: pack/transpose weights fp32 -> bf16  WxT/WhT [4][1024 n][1024 k]
//   Stage 2: Xp[m, g*1024+n] = x[m,:1024] @ Wx   (M=8192, N=4096, K=1024)
//   Stage 3: ONE persistent kernel, 64 WGs, 128 steps with device-wide
//            barrier; h slab staged via GLOBAL_LOAD_ASYNC_TO_LDS (DEV scope).
// ---------------------------------------------------------------------------

typedef __attribute__((ext_vector_type(16))) __bf16 bf16x16;
typedef __attribute__((ext_vector_type(8)))  float  floatx8;
typedef int v4i __attribute__((vector_size(16)));   // matches builtin proto
typedef unsigned short u16;

#define SIZE 1024            // hidden / input size
#define NB   64              // batch
#define NT   128             // time steps
#define NGATE 4
#define NTOT (NGATE * SIZE)  // 4096
#define MROWS (NT * NB)      // 8192
#define SEQ_BLOCKS 64        // persistent-kernel grid (must be co-resident)

#if defined(__has_builtin)
#if __has_builtin(__builtin_amdgcn_global_load_async_to_lds_b128) && \
    __has_builtin(__builtin_amdgcn_s_wait_asynccnt)
#define HAS_ASYNC_LDS 1
#endif
#endif
#ifndef HAS_ASYNC_LDS
#define HAS_ASYNC_LDS 0
#endif

__device__ __forceinline__ u16 f2bf(float f) {
  unsigned u = __float_as_uint(f);
  u += 0x7FFFu + ((u >> 16) & 1u);          // round-to-nearest-even
  return (u16)(u >> 16);
}

// load 16 contiguous bf16 (32 bytes) as a WMMA fragment (two b128 loads)
__device__ __forceinline__ bf16x16 ld_frag(const u16* p) {
  union { uint4 u[2]; bf16x16 v; } f;
  f.u[0] = *(const uint4*)(p);
  f.u[1] = *(const uint4*)(p + 8);
  return f.v;
}

// 16B global -> LDS copy.  Primary path: CDNA5 async-to-LDS with DEV scope
// (cpol scope=2 in bits [4:3]) so reads bypass the read-only WGP$ and hit L2,
// the coherence point for h written by other WGPs between steps.
__device__ __forceinline__ void stage_b128(u16* ldst, const u16* gsrc) {
#if HAS_ASYNC_LDS
  __builtin_amdgcn_global_load_async_to_lds_b128(
      (__attribute__((address_space(1))) v4i*)(u16*)gsrc,
      (__attribute__((address_space(3))) v4i*)ldst,
      0, /*cpol: scope=DEV*/ (2 << 3));
#else
  *(uint4*)ldst = *(const uint4*)gsrc;      // fallback (no DEV scope control)
#endif
}
__device__ __forceinline__ void stage_wait() {
#if HAS_ASYNC_LDS
  __builtin_amdgcn_s_wait_asynccnt(0);
#endif
}

#define WMMA_BF16(a, b, c) \
  __builtin_amdgcn_wmma_f32_16x16x32_bf16(false, (a), false, (b), (short)0, (c), false, false)

// ---------------------------------------------------------------------------
// Kernel 0: zero recurrent state + global barrier counter (ws is poisoned)
// ---------------------------------------------------------------------------
__global__ __launch_bounds__(256) void lstm_init(float* c, u16* h0, u16* h1,
                                                 unsigned* bar) {
  int i = blockIdx.x * 256 + threadIdx.x;   // grid covers 64*1024
  c[i]  = 0.0f;
  h0[i] = 0;
  h1[i] = 0;
  if (i == 0) *bar = 0u;
}

// ---------------------------------------------------------------------------
// Kernel 1: fp32 weights [2048,1024] (k-major) -> bf16 transposed [4][n][k]
// rows k<1024 -> WxT, k>=1024 -> WhT.   grid (32 n-tiles, 64 k-tiles, 4 gates)
// ---------------------------------------------------------------------------
__global__ __launch_bounds__(256) void lstm_pack_w(const float* __restrict__ wf,
                                                   const float* __restrict__ wu,
                                                   const float* __restrict__ wc,
                                                   const float* __restrict__ wo,
                                                   u16* __restrict__ WxT,
                                                   u16* __restrict__ WhT) {
  __shared__ u16 tile[32][33];
  const int g = blockIdx.z;
  const float* W = (g == 0) ? wf : (g == 1) ? wu : (g == 2) ? wc : wo;
  const int n0 = blockIdx.x * 32;
  const int k0 = blockIdx.y * 32;
  const int tx = threadIdx.x, ty = threadIdx.y;
#pragma unroll
  for (int j = 0; j < 4; ++j) {
    int kr = ty + j * 8;
    tile[kr][tx] = f2bf(W[(size_t)(k0 + kr) * SIZE + n0 + tx]);
  }
  __syncthreads();
  u16* dst = (k0 < SIZE) ? WxT : WhT;
  const int kk0 = k0 & (SIZE - 1);
#pragma unroll
  for (int j = 0; j < 4; ++j) {
    int nr = ty + j * 8;
    dst[(size_t)(g * SIZE + n0 + nr) * SIZE + kk0 + tx] = tile[tx][nr];
  }
}

// ---------------------------------------------------------------------------
// Kernel 2: Xp = x(bf16 on the fly) @ WxT.  Block tile 128x64, BK=32.
// 8 waves: wave tile 32x32 => 2x2 WMMA frags, f32 accum.
// Double-buffered A slab: one __syncthreads per K iteration.
// ---------------------------------------------------------------------------
#define BM 128
#define BN 64
#define BK 32
#define LDSA (BK + 8)   // 40 u16 = 80B rows: 16B aligned, conflict-staggered

__global__ __launch_bounds__(256) void lstm_xproj(const float* __restrict__ x,
                                                  const u16* __restrict__ WxT,
                                                  float* __restrict__ Xp) {
  __shared__ u16 sA[2][BM][LDSA];
  const int m0 = blockIdx.x * BM;
  const int n0 = blockIdx.y * BN;
  const int tid = threadIdx.x;
  const int lane = tid & 31, wave = tid >> 5;
  const int wr = wave >> 1;                 // 0..3 -> row offset wr*32
  const int wc = wave & 1;                  // 0..1 -> col offset wc*32
  const int mlane = lane & 15;
  const int khalf = (lane >> 4) * 16;       // lanes 16-31 hold upper K half

  floatx8 acc[2][2] = {{{0,0,0,0,0,0,0,0}, {0,0,0,0,0,0,0,0}},
                       {{0,0,0,0,0,0,0,0}, {0,0,0,0,0,0,0,0}}};

  const int lr = tid >> 3;                  // 0..31 staging row
  const int lk = (tid & 7) << 2;            // float4 column

  // stage A slab 128x32 (fp32 -> bf16) into buffer `buf`
  auto stage = [&](int buf, int k0) {
#pragma unroll
    for (int p = 0; p < 4; ++p) {
      int row = (p << 5) + lr;
      float4 v = *(const float4*)(x + (size_t)(m0 + row) * SIZE + k0 + lk);
      unsigned lo = ((unsigned)f2bf(v.y) << 16) | f2bf(v.x);
      unsigned hi = ((unsigned)f2bf(v.w) << 16) | f2bf(v.z);
      *(uint2*)&sA[buf][row][lk] = make_uint2(lo, hi);
    }
  };

  stage(0, 0);
  __syncthreads();

  for (int kb = 0; kb < SIZE / BK; ++kb) {
    const int cur = kb & 1;
    if (kb + 1 < SIZE / BK) stage(cur ^ 1, (kb + 1) * BK);

    bf16x16 a[2], b[2];
#pragma unroll
    for (int i = 0; i < 2; ++i)
      a[i] = ld_frag(&sA[cur][wr * 32 + i * 16 + mlane][khalf]);
#pragma unroll
    for (int j = 0; j < 2; ++j) {
      int n = n0 + wc * 32 + j * 16 + mlane;
      b[j] = ld_frag(WxT + (size_t)n * SIZE + kb * BK + khalf);
    }
#pragma unroll
    for (int i = 0; i < 2; ++i)
#pragma unroll
      for (int j = 0; j < 2; ++j)
        acc[i][j] = WMMA_BF16(a[i], b[j], acc[i][j]);
    __syncthreads();
  }

  const int mhi = (lane >> 4) * 8;          // C layout: lanes 16-31 -> M+8
#pragma unroll
  for (int i = 0; i < 2; ++i)
#pragma unroll
    for (int j = 0; j < 2; ++j)
#pragma unroll
      for (int r = 0; r < 8; ++r) {
        int row = m0 + wr * 32 + i * 16 + mhi + r;
        int col = n0 + wc * 32 + j * 16 + mlane;
        // streamed exactly once -> non-temporal, keep weights resident in L2
        __builtin_nontemporal_store(acc[i][j][r], Xp + (size_t)row * NTOT + col);
      }
}

// ---------------------------------------------------------------------------
// Kernel 3: persistent recurrent kernel.  64 WGs x 16 hidden columns,
// 128 steps with device-wide barrier (monotonic counter, release/acquire).
// 8 waves: wave = (gate g, row-half rh); each wave: 2 WMMA tiles 16x16.
// h double-buffered bf16; gate exchange via LDS; c/h update fused.
// ---------------------------------------------------------------------------
__global__ __launch_bounds__(256) void lstm_seq(const u16*   __restrict__ WhT,
                                                const float* __restrict__ Xp,
                                                const float* __restrict__ bf_,
                                                const float* __restrict__ bu_,
                                                const float* __restrict__ bc_,
                                                const float* __restrict__ bo_,
                                                u16*         __restrict__ h0,
                                                u16*         __restrict__ h1,
                                                float*       __restrict__ cst,
                                                float*       __restrict__ out,
                                                unsigned*    __restrict__ bar) {
  __shared__ u16   sH[2][NB][LDSA];         // double-buffered 64x32 h slab
  __shared__ float sG[NGATE][NB][16];       // activated gates for combine

  const int n0 = blockIdx.x * 16;
  const int tid = threadIdx.x;
  const int lane = tid & 31, wave = tid >> 5;
  const int g  = wave >> 1;                 // gate: 0=f 1=u 2=c 3=o
  const int rh = wave & 1;                  // row half (rows rh*32..+32)
  const int mlane = lane & 15;
  const int khalf = (lane >> 4) * 16;
  const int mhi   = (lane >> 4) * 8;
  const int srow = tid >> 2;                // 0..63 staging row
  const int skq  = (tid & 3) << 3;          // 0,8,16,24 (8 bf16 = 16B)
  const float* bias = (g == 0) ? bf_ : (g == 1) ? bu_ : (g == 2) ? bc_ : bo_;

  for (int t = 0; t < NT; ++t) {
    const u16* h_src = (t & 1) ? h1 : h0;
    u16*       h_dst = (t & 1) ? h0 : h1;

    floatx8 acc[2] = {{0,0,0,0,0,0,0,0}, {0,0,0,0,0,0,0,0}};

    // ------- K loop:  acc += h @ Wh  (double-buffered async h slab) -------
    stage_b128(&sH[0][srow][skq], h_src + (size_t)srow * SIZE + skq);
    stage_wait();
    __syncthreads();

    for (int kb = 0; kb < SIZE / BK; ++kb) {
      const int cur = kb & 1;
      if (kb + 1 < SIZE / BK)
        stage_b128(&sH[cur ^ 1][srow][skq],
                   h_src + (size_t)srow * SIZE + (kb + 1) * BK + skq);

      bf16x16 b = ld_frag(WhT + (size_t)(g * SIZE + n0 + mlane) * SIZE
                              + kb * BK + khalf);
#pragma unroll
      for (int i = 0; i < 2; ++i) {
        bf16x16 a = ld_frag(&sH[cur][rh * 32 + i * 16 + mlane][khalf]);
        acc[i] = WMMA_BF16(a, b, acc[i]);
      }
      stage_wait();
      __syncthreads();
    }

    // ------- epilogue: + x-projection + bias, activation, park in LDS -----
#pragma unroll
    for (int i = 0; i < 2; ++i)
#pragma unroll
      for (int r = 0; r < 8; ++r) {
        int row = rh * 32 + i * 16 + mhi + r;          // batch index
        int col = mlane;                               // local hidden column
        float v = acc[i][r]
                + __builtin_nontemporal_load(
                      Xp + (size_t)(t * NB + row) * NTOT + g * SIZE + n0 + col)
                + bias[n0 + col];
        v = (g == 2) ? tanhf(v) : (1.0f / (1.0f + __expf(-v)));
        sG[g][row][col] = v;
      }
    __syncthreads();

    // ------- combine gates: c' = c*f + u*g ; h = tanh(c')*o ---------------
#pragma unroll
    for (int p = 0; p < 4; ++p) {
      int e = tid + p * 256;                // 0..1023 -> (row, col)
      int row = e >> 4, col = e & 15;
      float f  = sG[0][row][col];
      float u  = sG[1][row][col];
      float gc = sG[2][row][col];
      float o  = sG[3][row][col];
      size_t ci = (size_t)row * SIZE + n0 + col;
      float cn = cst[ci] * f + u * gc;
      float h  = tanhf(cn) * o;
      cst[ci]   = cn;                        // column-owned: same WGP, coherent
      h_dst[ci] = f2bf(h);                   // write-through to L2
      __builtin_nontemporal_store(h, out + (size_t)(t * NB + row) * SIZE + n0 + col);
    }

    // ------- device-wide barrier between steps ----------------------------
    if (t + 1 < NT) {
      __threadfence();                       // drain stores to L2 (coherence pt)
      __syncthreads();
      if (tid == 0) {
        unsigned target = (unsigned)(t + 1) * (unsigned)gridDim.x;
        __hip_atomic_fetch_add(bar, 1u, __ATOMIC_RELEASE, __HIP_MEMORY_SCOPE_AGENT);
        while (__hip_atomic_load(bar, __ATOMIC_ACQUIRE, __HIP_MEMORY_SCOPE_AGENT)
               < target) {
          __builtin_amdgcn_s_sleep(2);
        }
      }
      __syncthreads();
    }
  }
}

// ---------------------------------------------------------------------------
// launcher
// ---------------------------------------------------------------------------
extern "C" void kernel_launch(void* const* d_in, const int* in_sizes, int n_in,
                              void* d_out, int out_size, void* d_ws, size_t ws_size,
                              hipStream_t stream) {
  const float* x  = (const float*)d_in[0];
  const float* wf = (const float*)d_in[1];
  const float* bf = (const float*)d_in[2];
  const float* wu = (const float*)d_in[3];
  const float* bu = (const float*)d_in[4];
  const float* wc = (const float*)d_in[5];
  const float* bc = (const float*)d_in[6];
  const float* wo = (const float*)d_in[7];
  const float* bo = (const float*)d_in[8];
  float* out = (float*)d_out;

  // workspace layout (~145 MB total)
  char* ws = (char*)d_ws;
  const size_t OFF_WXT = 0;                            // 4*1024*1024 bf16 = 8 MiB
  const size_t OFF_WHT = 8ull << 20;                   // 8 MiB
  const size_t OFF_H0  = 16ull << 20;                  // 64*1024 bf16 = 128 KiB
  const size_t OFF_H1  = OFF_H0 + (size_t)NB * SIZE * 2;
  const size_t OFF_C   = OFF_H1 + (size_t)NB * SIZE * 2;
  const size_t OFF_BAR = OFF_C  + (size_t)NB * SIZE * 4;
  const size_t OFF_XP  = OFF_BAR + 256;                // 8192*4096 f32 = 128 MiB
  u16*      WxT = (u16*)(ws + OFF_WXT);
  u16*      WhT = (u16*)(ws + OFF_WHT);
  u16*      h0  = (u16*)(ws + OFF_H0);
  u16*      h1  = (u16*)(ws + OFF_H1);
  float*    cst = (float*)(ws + OFF_C);
  unsigned* bar = (unsigned*)(ws + OFF_BAR);
  float*    Xp  = (float*)(ws + OFF_XP);

  // 0) zero recurrent state + barrier counter (deterministic across replays)
  lstm_init<<<(NB * SIZE) / 256, 256, 0, stream>>>(cst, h0, h1, bar);

  // 1) pack + transpose weights fp32 -> bf16 [gate][n][k]
  lstm_pack_w<<<dim3(SIZE / 32, 2 * SIZE / 32, NGATE), dim3(32, 8), 0, stream>>>(
      wf, wu, wc, wo, WxT, WhT);

  // 2) parallel x-projection GEMM: [8192,1024] @ [1024,4096]
  lstm_xproj<<<dim3(MROWS / BM, NTOT / BN), 256, 0, stream>>>(x, WxT, Xp);

  // 3) all 128 recurrent steps in ONE persistent kernel (64 WGs co-resident,
  //    device-wide barrier between steps; Wh stays hot in L2/WGP caches)
  lstm_seq<<<SEQ_BLOCKS, 256, 0, stream>>>(WhT, Xp, bf, bu, bc, bo,
                                           h0, h1, cst, out, bar);
}
